// CGCNN_82480551952828
// MI455X (gfx1250) — matrix operand order repaired and checked by reading, not compiled, not run on hardware
//
#include <hip/hip_runtime.h>
#include <stdint.h>

namespace {
constexpr int NN  = 50000;          // nodes
constexpr int NE  = 400000;         // edges (25000 * 16)
constexpr int DD  = 128;            // node dim
constexpr int DEA = 64;             // edge feature dim
constexpr int ZK  = 2 * DD + DEA;   // 320 concat dim
constexpr int KST = ZK / 32;        // 10 WMMA k-steps
constexpr int ZPAD_W = 164;         // padded z row stride in u32 (328 bf16 = 656B, bank-conflict free)
}

typedef __attribute__((ext_vector_type(16))) __bf16 v16bf;
typedef __attribute__((ext_vector_type(8)))  float  v8f;

__device__ __forceinline__ unsigned int f2bf_bits(float f) {
    unsigned int u = __float_as_uint(f);
    return (u + 0x7FFFu + ((u >> 16) & 1u)) >> 16;   // round-to-nearest-even
}
__device__ __forceinline__ unsigned int packbf(float a, float b) {
    return f2bf_bits(a) | (f2bf_bits(b) << 16);
}

// ---- transpose + convert one weight matrix: W [Z,D] f32 -> Wt [D,Z] bf16 ----
__global__ void wconv_kernel(const float* __restrict__ W, unsigned short* __restrict__ Wt) {
    int idx = blockIdx.x * blockDim.x + threadIdx.x;   // over D*Z
    if (idx >= DD * ZK) return;
    int n = idx / ZK;
    int k = idx - n * ZK;
    Wt[idx] = (unsigned short)f2bf_bits(W[k * DD + n]);
}

// ---- h1 = x (residual accumulator), hb = bf16(x) pairs ----
__global__ void prep_nodes_kernel(const float* __restrict__ x, float* __restrict__ h1,
                                  unsigned int* __restrict__ hb) {
    int idx = blockIdx.x * blockDim.x + threadIdx.x;   // over N*D/2 pairs
    if (idx >= NN * DD / 2) return;
    float a = x[2 * idx], b = x[2 * idx + 1];
    h1[2 * idx] = a;
    h1[2 * idx + 1] = b;
    hb[idx] = packbf(a, b);
}

// ---- r = relu(h1); h2 = r (residual accumulator for layer 2); hb = bf16(r) ----
__global__ void relu_prep_kernel(const float* __restrict__ h1, float* __restrict__ h2,
                                 unsigned int* __restrict__ hb) {
    int idx = blockIdx.x * blockDim.x + threadIdx.x;   // pairs
    if (idx >= NN * DD / 2) return;
    float a = h1[2 * idx], b = h1[2 * idx + 1];
    a = a > 0.f ? a : 0.f;
    b = b > 0.f ? b : 0.f;
    h2[2 * idx] = a;
    h2[2 * idx + 1] = b;
    hb[idx] = packbf(a, b);
}

// ---- edge message GEMM + scatter: 16 edges / block, 8 waves cover 128 cols ----
__global__ __launch_bounds__(256)
void edge_msg_kernel(const unsigned int* __restrict__ hb,        // [N, 64] bf16 pairs
                     const float*        __restrict__ edge_attr, // [E, 64] f32
                     const long long*    __restrict__ edge_index,// [2, E] i64
                     const unsigned short* __restrict__ WfT,     // [D, Z] bf16 (transposed)
                     const float*        __restrict__ bfv,       // [D]
                     const unsigned short* __restrict__ WsT,     // [D, Z] bf16
                     const float*        __restrict__ bsv,       // [D]
                     float*              __restrict__ hout)      // [N, D] accumulate (atomics)
{
    __shared__ __align__(16) unsigned int zs[16 * ZPAD_W];  // 16 edges x 328 bf16 (padded)
    __shared__ int dsts[16];

    const int tid = threadIdx.x;
    const int e0  = blockIdx.x * 16;

    if (tid < 16) dsts[tid] = (int)edge_index[NE + e0 + tid];

    // Stage z tile as bf16 pairs: [dst(64 u32) | src(64 u32) | edge_attr(32 u32)]
    for (int idx = tid; idx < 16 * 160; idx += 256) {
        int r = idx / 160;
        int w = idx - r * 160;
        int e = e0 + r;
        unsigned int val;
        if (w < 64) {
            int dn = (int)edge_index[NE + e];
            val = hb[dn * 64 + w];
        } else if (w < 128) {
            int sn = (int)edge_index[e];
            val = hb[sn * 64 + (w - 64)];
        } else {
            int c = (w - 128) * 2;
            val = packbf(edge_attr[e * 64 + c], edge_attr[e * 64 + c + 1]);
        }
        zs[r * ZPAD_W + w] = val;
    }
    __syncthreads();

    const int lane = tid & 31;
    const int wv   = tid >> 5;       // wave id -> output column tile
    const int c16  = lane & 15;      // A-row selector AND D-column selector
    const int hi   = lane >> 4;
    const int ncol = wv * 16 + c16;  // this lane's output column (0..127)

    v8f accf = {};
    v8f accs = {};

    const char* zbase = (const char*)zs + c16 * (ZPAD_W * 4);
    const unsigned short* wfp = WfT + ncol * ZK;
    const unsigned short* wsp = WsT + ncol * ZK;

    for (int ks = 0; ks < KST; ++ks) {
        // A fragment (16x32 bf16): lane holds row M=c16;
        // K = ks*32 + hi*16 .. +15 contiguous  -> two b128 LDS loads
        union { uint4 q[2]; v16bf v; } a;
        const char* ap = zbase + (ks * 32 + hi * 16) * 2;
        a.q[0] = *(const uint4*)(ap);
        a.q[1] = *(const uint4*)(ap + 16);

        // B fragments (32x16 bf16): lane holds N=c16, K = ks*32 + hi*16 .. +15 contiguous
        union { uint4 q[2]; v16bf v; } bF, bS;
        const unsigned short* fp = wfp + ks * 32 + hi * 16;
        bF.q[0] = *(const uint4*)(fp);
        bF.q[1] = *(const uint4*)(fp + 8);
        const unsigned short* sp = wsp + ks * 32 + hi * 16;
        bS.q[0] = *(const uint4*)(sp);
        bS.q[1] = *(const uint4*)(sp + 8);

        accf = __builtin_amdgcn_wmma_f32_16x16x32_bf16(false, a.v, false, bF.v,
                                                       (short)0, accf, false, false);
        accs = __builtin_amdgcn_wmma_f32_16x16x32_bf16(false, a.v, false, bS.v,
                                                       (short)0, accs, false, false);
    }

    const float biasF = bfv[ncol];
    const float biasS = bsv[ncol];

    #pragma unroll
    for (int v = 0; v < 8; ++v) {
        float g = accf[v] + biasF;
        float s = accs[v] + biasS;
        float gate = 1.0f / (1.0f + __expf(-g));
        float core = (s > 20.0f) ? s : log1pf(__expf(s));   // softplus
        float msg  = gate * core;
        int m = v + hi * 8;                 // edge row within tile (C/D layout)
        int dn = dsts[m];
        atomicAdd(&hout[dn * DD + ncol], msg);
    }
}

// ---- final linear D -> 1, one wave per node ----
__global__ __launch_bounds__(256)
void final_kernel(const float* __restrict__ h2, const float* __restrict__ Wfc,
                  const float* __restrict__ bfc, float* __restrict__ out) {
    int node = blockIdx.x * 8 + (threadIdx.x >> 5);
    int lane = threadIdx.x & 31;
    if (node >= NN) return;
    float s = 0.f;
    #pragma unroll
    for (int d = 0; d < DD; d += 32) s += h2[node * DD + d + lane] * Wfc[d + lane];
    #pragma unroll
    for (int off = 16; off > 0; off >>= 1) s += __shfl_xor(s, off, 32);
    if (lane == 0) out[node] = s + bfc[0];
}

extern "C" void kernel_launch(void* const* d_in, const int* in_sizes, int n_in,
                              void* d_out, int out_size, void* d_ws, size_t ws_size,
                              hipStream_t stream) {
    const float*     x   = (const float*)d_in[0];
    const long long* ei  = (const long long*)d_in[1];
    const float*     ea  = (const float*)d_in[2];
    const float*     Wf1 = (const float*)d_in[3];
    const float*     bf1 = (const float*)d_in[4];
    const float*     Ws1 = (const float*)d_in[5];
    const float*     bs1 = (const float*)d_in[6];
    const float*     Wf2 = (const float*)d_in[7];
    const float*     bf2 = (const float*)d_in[8];
    const float*     Ws2 = (const float*)d_in[9];
    const float*     bs2 = (const float*)d_in[10];
    const float*     Wfc = (const float*)d_in[11];
    const float*     bfc = (const float*)d_in[12];
    float*           out = (float*)d_out;

    // workspace layout (all sizes multiples of 256B)
    char* ws = (char*)d_ws;
    const size_t WT_BYTES = (size_t)DD * ZK * 2;       // 81920
    unsigned short* WfT1 = (unsigned short*)(ws);
    unsigned short* WsT1 = (unsigned short*)(ws + WT_BYTES);
    unsigned short* WfT2 = (unsigned short*)(ws + 2 * WT_BYTES);
    unsigned short* WsT2 = (unsigned short*)(ws + 3 * WT_BYTES);
    char* p = ws + 4 * WT_BYTES;
    unsigned int* hb = (unsigned int*)p;               p += (size_t)NN * DD / 2 * 4; // 12.8 MB
    float* h1 = (float*)p;                             p += (size_t)NN * DD * 4;     // 25.6 MB
    float* h2 = (float*)p;                                                           // 25.6 MB

    const int TB = 256;
    const int wgrid = (DD * ZK + TB - 1) / TB;         // 160
    wconv_kernel<<<wgrid, TB, 0, stream>>>(Wf1, WfT1);
    wconv_kernel<<<wgrid, TB, 0, stream>>>(Ws1, WsT1);
    wconv_kernel<<<wgrid, TB, 0, stream>>>(Wf2, WfT2);
    wconv_kernel<<<wgrid, TB, 0, stream>>>(Ws2, WsT2);

    const int ngrid = (NN * DD / 2 + TB - 1) / TB;     // 12500
    prep_nodes_kernel<<<ngrid, TB, 0, stream>>>(x, h1, hb);

    edge_msg_kernel<<<NE / 16, TB, 0, stream>>>(hb, ea, ei, WfT1, bf1, WsT1, bs1, h1);

    relu_prep_kernel<<<ngrid, TB, 0, stream>>>(h1, h2, hb);

    edge_msg_kernel<<<NE / 16, TB, 0, stream>>>(hb, ea, ei, WfT2, bf2, WsT2, bs2, h2);

    final_kernel<<<NN / 8, TB, 0, stream>>>(h2, Wfc, bfc, out);
}
